// Block_34196529610857
// MI455X (gfx1250) — matrix operand order repaired and checked by reading, not compiled, not run on hardware
//
#include <hip/hip_runtime.h>
#include <math.h>

#define DEVFN __device__ __forceinline__

typedef __attribute__((ext_vector_type(16))) __bf16 v16bf;
typedef __attribute__((ext_vector_type(8)))  __bf16 v8bf;
typedef __attribute__((ext_vector_type(8)))  float  v8f;
typedef __attribute__((ext_vector_type(4)))  unsigned int u32x4;
typedef __attribute__((ext_vector_type(8)))  unsigned int u32x8;

static constexpr int cB = 8, cN = 1024, cC = 1024, cH = 16, cD = 64, cHID = 4096;
static constexpr int cM = cB * cN;            // 8192 token rows
static constexpr float cSCALE = 0.125f;       // D^-0.5

// ---------------- helpers ----------------

DEVFN unsigned short f2bf(float f) {
  unsigned u = __float_as_uint(f);
  u += 0x7FFFu + ((u >> 16) & 1u);            // round-to-nearest-even
  return (unsigned short)(u >> 16);
}

// A-operand (16x32, bf16) contiguous k order (same as load_b_frag):
// lanes 0-15 : M=lane,     elems 0..15 = K[0..15]
// lanes 16-31: M=lane-16,  elems 0..15 = K[16..31]
DEVFN v16bf load_a_frag(const unsigned short* rowp, int kbase, int lane) {
  int g = (lane >> 4) * 16;
  v8bf lo = *(const v8bf*)(rowp + kbase + g);
  v8bf hi = *(const v8bf*)(rowp + kbase + 8 + g);
  v16bf a;
#pragma unroll
  for (int i = 0; i < 8; ++i) { a[i] = lo[i]; a[i + 8] = hi[i]; }
  return a;
}

// B-operand (32x16, bf16): lane n = col, 16 contiguous K starting at (lane/16)*16
DEVFN v16bf load_b_frag(const unsigned short* rowp, int kbase, int lane) {
  return *(const v16bf*)(rowp + kbase + (lane >> 4) * 16);
}

DEVFN v8f wmma_bf16(v16bf a, v16bf b, v8f c) {
  return __builtin_amdgcn_wmma_f32_16x16x32_bf16(false, a, false, b, (short)0, c,
                                                 false, false);
}

DEVFN v8f zero8() {
  v8f z;
#pragma unroll
  for (int i = 0; i < 8; ++i) z[i] = 0.0f;
  return z;
}

DEVFN float gelu_exact(float x) {
  return 0.5f * x * (1.0f + erff(x * 0.70710678118654752f));
}

// ---- Tensor Data Mover: 2D tile (16 rows x 64 bf16 cols) global -> LDS ----
// All descriptor words are wave-uniform (forced via readfirstlane).
DEVFN void tdm_load_2d(unsigned ldsOff, const void* gptr, unsigned K_elems) {
  unsigned long long ga = (unsigned long long)gptr;
  u32x4 g0;
  g0[0] = 1u;                                              // count=1, load descriptor
  g0[1] = (unsigned)__builtin_amdgcn_readfirstlane((int)ldsOff);
  g0[2] = (unsigned)__builtin_amdgcn_readfirstlane((int)(unsigned)(ga & 0xFFFFFFFFu));
  g0[3] = (unsigned)__builtin_amdgcn_readfirstlane(
              (int)((unsigned)((ga >> 32) & 0x01FFFFFFu) | 0x80000000u)); // type=2
  unsigned T0 = K_elems;                                   // tensor_dim0 (row length)
  unsigned T1 = 1u << 20;                                  // tensor_dim1 (large)
  u32x8 g1;
  g1[0] = 1u << 16;                                        // data_size=2B, mask=0
  g1[1] = (T0 & 0xFFFFu) << 16;                            // tensor_dim0 lo
  g1[2] = (T0 >> 16) | ((T1 & 0xFFFFu) << 16);             // dim0 hi | dim1 lo
  g1[3] = (T1 >> 16) | (64u << 16);                        // dim1 hi | tile_dim0=64
  g1[4] = 16u;                                             // tile_dim1=16, tile_dim2=0
  g1[5] = K_elems;                                         // tensor_dim0_stride lo32
  g1[6] = 0u;                                              // stride hi | dim1_stride lo
  g1[7] = 0u;
  asm volatile("tensor_load_to_lds %0, %1" : : "s"(g0), "s"(g1) : "memory");
}

// ---------------- f32 -> bf16 weight conversion ----------------

__global__ __launch_bounds__(256) void cvt_bf16_kernel(const float* __restrict__ src,
                                                       unsigned short* __restrict__ dst,
                                                       int n) {
  int i = blockIdx.x * 256 + threadIdx.x;
  if (i < n) dst[i] = f2bf(src[i]);
}

// ---------------- LayerNorm (block per row) ----------------

__global__ __launch_bounds__(256) void ln_bf16_kernel(const float* __restrict__ X,
                                                      const float* __restrict__ g,
                                                      const float* __restrict__ bta,
                                                      unsigned short* __restrict__ out) {
  __shared__ float ss[256], sq[256];
  int row = blockIdx.x, tid = threadIdx.x;
  const float* xr = X + (size_t)row * cC;
  float v[4], s = 0.0f, q = 0.0f;
#pragma unroll
  for (int i = 0; i < 4; ++i) {
    v[i] = xr[tid + i * 256];
    s += v[i]; q += v[i] * v[i];
  }
  ss[tid] = s; sq[tid] = q;
  __syncthreads();
  for (int off = 128; off > 0; off >>= 1) {
    if (tid < off) { ss[tid] += ss[tid + off]; sq[tid] += sq[tid + off]; }
    __syncthreads();
  }
  float mean = ss[0] * (1.0f / cC);
  float var  = sq[0] * (1.0f / cC) - mean * mean;
  float inv  = rsqrtf(var + 1e-5f);
#pragma unroll
  for (int i = 0; i < 4; ++i) {
    int c = tid + i * 256;
    out[(size_t)row * cC + c] = f2bf((v[i] - mean) * inv * g[c] + bta[c]);
  }
}

// ---------------- bf16 WMMA GEMM with TDM-staged LDS tiles ----------------
// out[m,n] = sum_k A[m,k]*W[n,k].  Block: 256 thr = 8 waves as 4(M) x 2(N).
// Block tile 128x128; k-step 64; each wave DMAs a 16x64 slice of A and B into
// LDS via tensor_load_to_lds, double buffered (DMA of step k+1 overlaps compute).
// MODE 0: scatter into q[B,H,N,D], k[B,H,N,D], vT[B,H,D,N]   (bf16, no bias)
// MODE 1: outF = resid + acc + bias                           (f32)
// MODE 2: outQ = bf16(gelu(acc + bias)), row stride = Nout
// MODE 3: outF = outF + acc + bias                            (f32, in-place residual)

template <int MODE>
__global__ __launch_bounds__(256) void gemm_bf16_kernel(
    const unsigned short* __restrict__ A, const unsigned short* __restrict__ W,
    int Nout, int K,
    const float* __restrict__ bias, const float* __restrict__ resid,
    float* outF,
    unsigned short* __restrict__ outQ, unsigned short* __restrict__ outK,
    unsigned short* __restrict__ outV) {
  __shared__ __attribute__((aligned(16))) unsigned short Atile[2][128][64];
  __shared__ __attribute__((aligned(16))) unsigned short Btile[2][128][64];

  int tid = threadIdx.x, lane = tid & 31, wave = tid >> 5;
  int wm = wave >> 1, wn = wave & 1;
  int m0b = blockIdx.x * 128, n0b = blockIdx.y * 128;

  const unsigned short* aSrc = A + (size_t)(m0b + wave * 16) * K;
  const unsigned short* bSrc = W + (size_t)(n0b + wave * 16) * K;
  unsigned ldsA[2], ldsB[2];
  ldsA[0] = (unsigned)(unsigned long long)&Atile[0][wave * 16][0];
  ldsA[1] = (unsigned)(unsigned long long)&Atile[1][wave * 16][0];
  ldsB[0] = (unsigned)(unsigned long long)&Btile[0][wave * 16][0];
  ldsB[1] = (unsigned)(unsigned long long)&Btile[1][wave * 16][0];

  v8f acc[2][4];
#pragma unroll
  for (int i = 0; i < 2; ++i)
#pragma unroll
    for (int j = 0; j < 4; ++j) acc[i][j] = zero8();

  tdm_load_2d(ldsA[0], aSrc, (unsigned)K);
  tdm_load_2d(ldsB[0], bSrc, (unsigned)K);

  int buf = 0;
  for (int kk = 0; kk < K; kk += 64) {
    __builtin_amdgcn_s_wait_tensorcnt(0);   // my slice landed in LDS
    __syncthreads();                        // everyone's slice landed; prev reads done
    if (kk + 64 < K) {
      tdm_load_2d(ldsA[buf ^ 1], aSrc + kk + 64, (unsigned)K);
      tdm_load_2d(ldsB[buf ^ 1], bSrc + kk + 64, (unsigned)K);
    }
#pragma unroll
    for (int kt = 0; kt < 2; ++kt) {
      v16bf af[2], bfv[4];
#pragma unroll
      for (int i = 0; i < 2; ++i)
        af[i] = load_a_frag(&Atile[buf][wm * 32 + i * 16 + (lane & 15)][0], kt * 32, lane);
#pragma unroll
      for (int j = 0; j < 4; ++j)
        bfv[j] = load_b_frag(&Btile[buf][wn * 64 + j * 16 + (lane & 15)][0], kt * 32, lane);
#pragma unroll
      for (int i = 0; i < 2; ++i)
#pragma unroll
        for (int j = 0; j < 4; ++j) acc[i][j] = wmma_bf16(af[i], bfv[j], acc[i][j]);
    }
    buf ^= 1;
  }

  int m0 = m0b + wm * 32;
  int n0 = n0b + wn * 64;
  int rowoff = (lane < 16) ? 0 : 8;
  int colo = lane & 15;
#pragma unroll
  for (int i = 0; i < 2; ++i)
#pragma unroll
    for (int j = 0; j < 4; ++j)
#pragma unroll
      for (int r = 0; r < 8; ++r) {
        int row = m0 + i * 16 + r + rowoff;
        int col = n0 + j * 16 + colo;
        float v = acc[i][j][r];
        if (MODE == 0) {
          int sel = col >> 10;          // 0:q 1:k 2:v
          int h = (col >> 6) & 15;
          int d = col & 63;
          int b = row >> 10;
          int n = row & 1023;
          unsigned short hv = f2bf(v);
          size_t bh = (size_t)(b * cH + h);
          if (sel == 0)      outQ[(bh * cN + n) * cD + d] = hv;
          else if (sel == 1) outK[(bh * cN + n) * cD + d] = hv;
          else               outV[(bh * cD + d) * cN + n] = hv;
        } else if (MODE == 1) {
          size_t idx = (size_t)row * cC + col;
          outF[idx] = resid[idx] + v + bias[col];
        } else if (MODE == 2) {
          outQ[(size_t)row * Nout + col] = f2bf(gelu_exact(v + bias[col]));
        } else {
          size_t idx = (size_t)row * cC + col;
          outF[idx] = outF[idx] + v + bias[col];
        }
      }
}

// ---------------- flash attention: wave per 16-query tile ----------------
// Softmax without max-subtraction (shift-invariant; logits are O(1) here and
// clamped at 30 so fp32 exp cannot overflow; masked keys give exp(-1e4)=0).
// Row sums accumulated per lane, reduced across the 16-lane half once at end.

__global__ __launch_bounds__(256) void attn_kernel(
    const unsigned short* __restrict__ Q, const unsigned short* __restrict__ Kt,
    const unsigned short* __restrict__ Vt, const int* __restrict__ length,
    unsigned short* __restrict__ Y) {
  __shared__ unsigned short ldsP[8][16][32];   // per-wave P tile (16 q x 32 keys)
  int tid = threadIdx.x, lane = tid & 31, w = tid >> 5;
  int qtile = blockIdx.x * 8 + w;
  int bh = qtile >> 6;                          // 64 q-tiles per (b,h)
  int qt = qtile & 63;
  int b = bh >> 4;
  int h = bh & 15;
  int q0 = qt * 16;

  const unsigned short* qbase = Q + (size_t)bh * cN * cD;
  const unsigned short* kbase = Kt + (size_t)bh * cN * cD;
  const unsigned short* vbase = Vt + (size_t)bh * cD * cN;
  int len = length[b];

  const unsigned short* qrow = qbase + (size_t)(q0 + (lane & 15)) * cD;
  v16bf aq0 = load_a_frag(qrow, 0, lane);       // d 0..31
  v16bf aq1 = load_a_frag(qrow, 32, lane);      // d 32..63

  v8f acc[4];
#pragma unroll
  for (int dt = 0; dt < 4; ++dt) acc[dt] = zero8();
  float lsum[8];
#pragma unroll
  for (int r = 0; r < 8; ++r) lsum[r] = 0.0f;

  for (int kk = 0; kk < cN; kk += 32) {
    // ---- S = q . k^T for two 16-key tiles (4 WMMAs) ----
    const unsigned short* krow0 = kbase + (size_t)(kk + (lane & 15)) * cD;
    const unsigned short* krow1 = kbase + (size_t)(kk + 16 + (lane & 15)) * cD;
    v8f s0 = zero8(), s1 = zero8();
    s0 = wmma_bf16(aq0, load_b_frag(krow0, 0, lane), s0);
    s0 = wmma_bf16(aq1, load_b_frag(krow0, 32, lane), s0);
    s1 = wmma_bf16(aq0, load_b_frag(krow1, 0, lane), s1);
    s1 = wmma_bf16(aq1, load_b_frag(krow1, 32, lane), s1);

    int j0 = kk + (lane & 15);
    float mask0 = ((j0) >= len) ? -10000.0f : 0.0f;
    float mask1 = ((j0 + 16) >= len) ? -10000.0f : 0.0f;

    // ---- P = exp(S*scale + mask), per-lane partial row sums, no shuffles ----
#pragma unroll
    for (int r = 0; r < 8; ++r) {
      float p0 = __expf(fminf(s0[r] * cSCALE + mask0, 30.0f));
      float p1 = __expf(fminf(s1[r] * cSCALE + mask1, 30.0f));
      lsum[r] += p0 + p1;
      int prow = r + ((lane < 16) ? 0 : 8);
      ldsP[w][prow][lane & 15] = f2bf(p0);
      ldsP[w][prow][16 + (lane & 15)] = f2bf(p1);
    }
    // intra-wave LDS RAW: compiler inserts s_wait_dscnt; LDS region is per-wave

    // ---- O += P . V  (V transposed [d][key]: contiguous keys, 4 WMMAs) ----
    v16bf pA = load_a_frag(&ldsP[w][lane & 15][0], 0, lane);
#pragma unroll
    for (int dt = 0; dt < 4; ++dt) {
      const unsigned short* vrow =
          vbase + (size_t)(dt * 16 + (lane & 15)) * cN + kk + (lane >> 4) * 16;
      v16bf bv = *(const v16bf*)vrow;
      acc[dt] = wmma_bf16(pA, bv, acc[dt]);
    }
  }

  // ---- one cross-lane reduction of the row sums, then normalize + store ----
  float li[8];
#pragma unroll
  for (int r = 0; r < 8; ++r) {
    float rs = lsum[r];
#pragma unroll
    for (int o = 8; o >= 1; o >>= 1) rs += __shfl_xor(rs, o, 32);
    li[r] = rs;
  }
#pragma unroll
  for (int dt = 0; dt < 4; ++dt)
#pragma unroll
    for (int r = 0; r < 8; ++r) {
      int row = q0 + r + ((lane < 16) ? 0 : 8);
      int d = dt * 16 + (lane & 15);
      Y[((size_t)b * cN + row) * cC + h * cD + d] = f2bf(acc[dt][r] / li[r]);
    }
}

// ---------------- host side ----------------

extern "C" void kernel_launch(void* const* d_in, const int* in_sizes, int n_in,
                              void* d_out, int out_size, void* d_ws, size_t ws_size,
                              hipStream_t stream) {
  const float* x     = (const float*)d_in[0];
  const int*   len   = (const int*)d_in[1];
  const float* g1    = (const float*)d_in[2];
  const float* b1    = (const float*)d_in[3];
  const float* Wqkv  = (const float*)d_in[4];
  const float* Wproj = (const float*)d_in[5];
  const float* bproj = (const float*)d_in[6];
  const float* g2    = (const float*)d_in[7];
  const float* b2    = (const float*)d_in[8];
  const float* W1    = (const float*)d_in[9];
  const float* bb1   = (const float*)d_in[10];
  const float* W2    = (const float*)d_in[11];
  const float* bb2   = (const float*)d_in[12];
  float* out = (float*)d_out;
  unsigned char* ws = (unsigned char*)d_ws;

  // workspace layout (bytes)
  const size_t OFF_WQKV  = 0;                           // 3C*C bf16 = 6 MB
  const size_t OFF_WPROJ = 6291456;                     // C*C  bf16 = 2 MB
  const size_t OFF_W1    = 8388608;                     // HID*C bf16 = 8 MB
  const size_t OFF_W2    = 16777216;                    // C*HID bf16 = 8 MB
  const size_t OFF_XN    = 25165824;                    // M*C bf16 = 16 MB
  const size_t OFF_Q     = 41943040;                    // 16 MB
  const size_t OFF_K     = 58720256;                    // 16 MB
  const size_t OFF_V     = 75497472;                    // 16 MB (transposed)
  const size_t OFF_Y     = 92274688;                    // 16 MB
  const size_t OFF_H     = OFF_Q;                       // 64 MB, reuses dead q/k/v/y

  unsigned short* wqkv_bf  = (unsigned short*)(ws + OFF_WQKV);
  unsigned short* wproj_bf = (unsigned short*)(ws + OFF_WPROJ);
  unsigned short* w1_bf    = (unsigned short*)(ws + OFF_W1);
  unsigned short* w2_bf    = (unsigned short*)(ws + OFF_W2);
  unsigned short* xn_bf    = (unsigned short*)(ws + OFF_XN);
  unsigned short* q_bf     = (unsigned short*)(ws + OFF_Q);
  unsigned short* k_bf     = (unsigned short*)(ws + OFF_K);
  unsigned short* v_bf     = (unsigned short*)(ws + OFF_V);
  unsigned short* y_bf     = (unsigned short*)(ws + OFF_Y);
  unsigned short* h_bf     = (unsigned short*)(ws + OFF_H);

  // 1) weights -> bf16
  cvt_bf16_kernel<<<(3 * cC * cC) / 256, 256, 0, stream>>>(Wqkv, wqkv_bf, 3 * cC * cC);
  cvt_bf16_kernel<<<(cC * cC) / 256, 256, 0, stream>>>(Wproj, wproj_bf, cC * cC);
  cvt_bf16_kernel<<<(cHID * cC) / 256, 256, 0, stream>>>(W1, w1_bf, cHID * cC);
  cvt_bf16_kernel<<<(cC * cHID) / 256, 256, 0, stream>>>(W2, w2_bf, cC * cHID);

  // 2) LN1 -> bf16
  ln_bf16_kernel<<<cM, 256, 0, stream>>>(x, g1, b1, xn_bf);

  // 3) QKV GEMM: [8192,1024] x [3072,1024]^T, scatter q/k/vT
  gemm_bf16_kernel<0><<<dim3(cM / 128, (3 * cC) / 128), 256, 0, stream>>>(
      xn_bf, wqkv_bf, 3 * cC, cC, nullptr, nullptr, nullptr, q_bf, k_bf, v_bf);

  // 4) flash attention -> y bf16
  attn_kernel<<<(cB * cH * (cN / 16)) / 8, 256, 0, stream>>>(q_bf, k_bf, v_bf, len, y_bf);

  // 5) proj GEMM + bias + residual -> out f32
  gemm_bf16_kernel<1><<<dim3(cM / 128, cC / 128), 256, 0, stream>>>(
      y_bf, wproj_bf, cC, cC, bproj, x, out, nullptr, nullptr, nullptr);

  // 6) LN2 -> bf16
  ln_bf16_kernel<<<cM, 256, 0, stream>>>(out, g2, b2, xn_bf);

  // 7) MLP1 GEMM + bias + exact GELU -> h bf16
  gemm_bf16_kernel<2><<<dim3(cM / 128, cHID / 128), 256, 0, stream>>>(
      xn_bf, w1_bf, cHID, cC, bb1, nullptr, nullptr, h_bf, nullptr, nullptr);

  // 8) MLP2 GEMM + bias + residual accumulate -> out f32
  gemm_bf16_kernel<3><<<dim3(cM / 128, cC / 128), 256, 0, stream>>>(
      h_bf, w2_bf, cC, cHID, bb2, nullptr, out, nullptr, nullptr, nullptr);
}
